// RewardPredictor_56599079027190
// MI455X (gfx1250) — matrix operand order repaired and checked by reading, not compiled, not run on hardware
//
#include <hip/hip_runtime.h>
#include <hip/hip_bf16.h>

// ---------------------------------------------------------------------------
// MI455X (gfx1250, wave32) implementation of the RewardPredictor forward pass.
// GEMM-shaped work -> v_wmma_f32_16x16x32_f16 (one wave per 16x16 tile).
// Small-channel convs + BN stats + LSTM pointwise -> f32 VALU.
// Activations/weights staged as f16 to halve HBM traffic (memory-bound model).
// ---------------------------------------------------------------------------

typedef __attribute__((ext_vector_type(16))) _Float16 v16h;
typedef __attribute__((ext_vector_type(8)))  float    v8f;

__device__ __forceinline__ float sigmf(float v) { return 1.0f / (1.0f + expf(-v)); }

// ---------------- weight preparation ----------------

__global__ void k_cast_f16(const float* __restrict__ s, _Float16* __restrict__ d, int n) {
  int i = blockIdx.x * blockDim.x + threadIdx.x;
  if (i < n) d[i] = (_Float16)s[i];
}

// OIHW f32 -> [Co][Kp] f16 with k = (ky*3+kx)*Ci + ci, zero-padded to Kp
__global__ void k_repack_conv_w(const float* __restrict__ w, _Float16* __restrict__ out,
                                int Ci, int Co, int Kp) {
  int idx = blockIdx.x * blockDim.x + threadIdx.x;
  if (idx >= Co * Kp) return;
  int k = idx % Kp, o = idx / Kp;
  float v = 0.0f;
  if (k < 9 * Ci) { int ci = k % Ci, tap = k / Ci; v = w[(o * Ci + ci) * 9 + tap]; }
  out[idx] = (_Float16)v;
}

// conv_f on 1x1 spatial: only center tap (ky=kx=1 -> tap 4) contributes
__global__ void k_repack_convf(const float* __restrict__ w, _Float16* __restrict__ out) {
  int i = blockIdx.x * blockDim.x + threadIdx.x;
  if (i >= 128 * 128) return;
  int o = i >> 7, ci = i & 127;
  out[i] = (_Float16)w[(o * 128 + ci) * 9 + 4];
}

// Wcat = [Wih | Whh] as [128][64] f16 ; bsum = bih + bhh
__global__ void k_pack_lstm(const float* __restrict__ wih, const float* __restrict__ whh,
                            const float* __restrict__ bih, const float* __restrict__ bhh,
                            _Float16* __restrict__ wcat, float* __restrict__ bsum) {
  int i = blockIdx.x * blockDim.x + threadIdx.x;
  if (i < 128 * 64) {
    int o = i >> 6, k = i & 63;
    float v = (k < 32) ? wih[o * 32 + k] : whh[o * 32 + (k - 32)];
    wcat[i] = (_Float16)v;
  }
  if (i < 128) bsum[i] = bih[i] + bhh[i];
}

// ---------------- direct convs (channels too narrow for 16x16 tiles) --------

// layer1: x[b][t][ci][64][64] f32 (NCHW, frame t) -> pre NHWC f32 [512,64,64,4]
__global__ void k_conv1(const float* __restrict__ x, const float* __restrict__ w,
                        const float* __restrict__ b, float* __restrict__ pre, int t) {
  long long idx = (long long)blockIdx.x * blockDim.x + threadIdx.x;
  if (idx >= 512LL * 64 * 64) return;
  int xx = idx & 63, yy = (idx >> 6) & 63, bb = (int)(idx >> 12);
  const float* xb = x + (size_t)(bb * 3 + t) * 3 * 4096;
  float acc[4];
#pragma unroll
  for (int o = 0; o < 4; ++o) acc[o] = b[o];
  for (int ci = 0; ci < 3; ++ci)
    for (int ky = -1; ky <= 1; ++ky) {
      int y2 = yy + ky; if ((unsigned)y2 >= 64u) continue;
      for (int kx = -1; kx <= 1; ++kx) {
        int x2 = xx + kx; if ((unsigned)x2 >= 64u) continue;
        float v = xb[ci * 4096 + y2 * 64 + x2];
        int tap = (ky + 1) * 3 + (kx + 1);
#pragma unroll
        for (int o = 0; o < 4; ++o) acc[o] += v * w[(o * 3 + ci) * 9 + tap];
      }
    }
  float* op = pre + (size_t)idx * 4;
#pragma unroll
  for (int o = 0; o < 4; ++o) op[o] = acc[o];
}

// generic NHWC direct conv (used for layer2: Ci=4, Co=8)
__global__ void k_conv_direct(const _Float16* __restrict__ in, const float* __restrict__ w,
                              const float* __restrict__ bias, float* __restrict__ pre,
                              int B, int H, int W, int Ci, int Co) {
  long long idx = (long long)blockIdx.x * blockDim.x + threadIdx.x;
  long long total = (long long)B * H * W;
  if (idx >= total) return;
  int x = (int)(idx % W);
  long long r = idx / W;
  int y = (int)(r % H), b = (int)(r / H);
  float acc[8];
  for (int o = 0; o < Co; ++o) acc[o] = bias[o];
  for (int ky = -1; ky <= 1; ++ky) {
    int yy = y + ky; if ((unsigned)yy >= (unsigned)H) continue;
    for (int kx = -1; kx <= 1; ++kx) {
      int xx = x + kx; if ((unsigned)xx >= (unsigned)W) continue;
      int tap = (ky + 1) * 3 + (kx + 1);
      const _Float16* ip = in + ((size_t)(b * H + yy) * W + xx) * Ci;
      for (int ci = 0; ci < Ci; ++ci) {
        float v = (float)ip[ci];
        for (int o = 0; o < Co; ++o) acc[o] += v * w[(o * Ci + ci) * 9 + tap];
      }
    }
  }
  float* op = pre + (size_t)idx * Co;
  for (int o = 0; o < Co; ++o) op[o] = acc[o];
}

// ---------------- im2col + WMMA GEMM ----------------------------------------

__global__ void k_im2col3x3(const _Float16* __restrict__ act, _Float16* __restrict__ cols,
                            int B, int H, int W, int Ci, int Kp) {
  long long idx = (long long)blockIdx.x * blockDim.x + threadIdx.x;
  long long total = (long long)B * H * W * Kp;
  if (idx >= total) return;
  int k = (int)(idx % Kp);
  long long r = idx / Kp;
  int x = (int)(r % W); r /= W;
  int y = (int)(r % H);
  int b = (int)(r / H);
  _Float16 v = (_Float16)0.0f;
  if (k < 9 * Ci) {
    int ci = k % Ci, tap = k / Ci;
    int yy = y + tap / 3 - 1, xx = x + tap % 3 - 1;
    if ((unsigned)yy < (unsigned)H && (unsigned)xx < (unsigned)W)
      v = act[((size_t)(b * H + yy) * W + xx) * Ci + ci];
  }
  cols[idx] = v;
}

// Y[M,N] = A[M,K] * Wt[N,K]^T + bias ; one wave per 16x16 tile, K steps of 32.
// Wave32 fragment layouts per CDNA5 ISA 7.12.2:
//   A: lane holds row M=lane&15; lo half-wave K {0..7,16..23}, hi {8..15,24..31}
//   B: lane holds col N=lane&15; lo half-wave K {0..7,16..23}, hi {8..15,24..31}
//   D: VGPR r -> row r (lanes 0-15) / row r+8 (lanes 16-31), col = lane&15
__global__ void k_gemm_wmma(const _Float16* __restrict__ A, const _Float16* __restrict__ Wt,
                            const float* __restrict__ bias, float* __restrict__ Y,
                            int M, int N, int K, int tiles) {
  int wave = blockIdx.x * (blockDim.x >> 5) + ((int)threadIdx.x >> 5);
  if (wave >= tiles) return;
  int lane = threadIdx.x & 31;
  int tilesN = N >> 4;
  int tM = (wave / tilesN) << 4;
  int tN = (wave % tilesN) << 4;
  int hi = lane >> 4;
  int l15 = lane & 15;
  const _Float16* arow = A + (size_t)(tM + l15) * K;
  const _Float16* wrow = Wt + (size_t)(tN + l15) * K;
  v8f acc = {};
  for (int k0 = 0; k0 < K; k0 += 32) {
    v16h af, bf;
    int ka = k0 + (hi ? 8 : 0);
#pragma unroll
    for (int i = 0; i < 8; ++i) { af[i] = arow[ka + i]; af[i + 8] = arow[ka + 16 + i]; }
    int kb = k0 + (hi ? 8 : 0);
#pragma unroll
    for (int i = 0; i < 8; ++i) { bf[i] = wrow[kb + i]; bf[i + 8] = wrow[kb + 16 + i]; }
    acc = __builtin_amdgcn_wmma_f32_16x16x32_f16(false, af, false, bf, (short)0, acc,
                                                 false, false);
  }
  int n = tN + l15;
  float bv = bias ? bias[n] : 0.0f;
#pragma unroll
  for (int r = 0; r < 8; ++r) {
    int m = tM + r + (hi ? 8 : 0);
    Y[(size_t)m * N + n] = acc[r] + bv;
  }
}

// ---------------- batchnorm (train mode, biased variance) -------------------

__global__ void k_bn_stats(const float* __restrict__ pre, float* __restrict__ mean,
                           float* __restrict__ istd, int C, int Ncnt) {
  int c = blockIdx.x;
  float s = 0.0f, q = 0.0f;
  for (long long i = threadIdx.x; i < Ncnt; i += blockDim.x) {
    float v = pre[i * C + c];
    s += v; q += v * v;
  }
  __shared__ float ss[256], sq[256];
  int tid = threadIdx.x;
  ss[tid] = s; sq[tid] = q;
  __syncthreads();
  for (int o = 128; o > 0; o >>= 1) {
    if (tid < o) { ss[tid] += ss[tid + o]; sq[tid] += sq[tid + o]; }
    __syncthreads();
  }
  if (tid == 0) {
    float m = ss[0] / (float)Ncnt;
    float var = sq[0] / (float)Ncnt - m * m;
    mean[c] = m;
    istd[c] = rsqrtf(var + 1e-5f);
  }
}

// normalize + relu + 2x2 maxpool (NHWC), out f16 at half spatial
__global__ void k_bn_relu_pool(const float* __restrict__ pre, const float* __restrict__ mean,
                               const float* __restrict__ istd, const float* __restrict__ gam,
                               const float* __restrict__ bet, _Float16* __restrict__ out,
                               int B, int H, int W, int C) {
  int H2 = H >> 1, W2 = W >> 1;
  long long total = (long long)B * H2 * W2 * C;
  long long idx = (long long)blockIdx.x * blockDim.x + threadIdx.x;
  if (idx >= total) return;
  int c = (int)(idx % C);
  long long r = idx / C;
  int x2 = (int)(r % W2); r /= W2;
  int y2 = (int)(r % H2);
  int b = (int)(r / H2);
  float mu = mean[c], is = istd[c], g = gam[c], be = bet[c];
  const float* p = pre + (((size_t)(b * H + 2 * y2) * W + 2 * x2) * C + c);
  size_t rowstr = (size_t)W * C;
  float a0 = (p[0] - mu) * is * g + be;
  float a1 = (p[C] - mu) * is * g + be;
  float a2 = (p[rowstr] - mu) * is * g + be;
  float a3 = (p[rowstr + C] - mu) * is * g + be;
  float m = fmaxf(fmaxf(a0, a1), fmaxf(a2, a3));
  out[idx] = (_Float16)fmaxf(m, 0.0f);
}

__global__ void k_bn_relu(const float* __restrict__ pre, const float* __restrict__ mean,
                          const float* __restrict__ istd, const float* __restrict__ gam,
                          const float* __restrict__ bet, _Float16* __restrict__ out,
                          int C, long long total) {
  long long idx = (long long)blockIdx.x * blockDim.x + threadIdx.x;
  if (idx >= total) return;
  int c = (int)(idx % C);
  float v = (pre[idx] - mean[c]) * istd[c] * gam[c] + bet[c];
  out[idx] = (_Float16)fmaxf(v, 0.0f);
}

// ---------------- glue / LSTM -----------------------------------------------

__global__ void k_cast_scatter(const float* __restrict__ src, _Float16* __restrict__ dst,
                               int M, int N, int ld, int off) {
  int i = blockIdx.x * blockDim.x + threadIdx.x;
  if (i >= M * N) return;
  int m = i / N, n = i % N;
  dst[(size_t)m * ld + off + n] = (_Float16)src[i];
}

// s[:, :32] = z (LSTM input), s[:, 32:] = h0 = 0 ; c0 = 0
__global__ void k_lstm_init(const float* __restrict__ z, _Float16* __restrict__ s,
                            float* __restrict__ c, int B) {
  int i = blockIdx.x * blockDim.x + threadIdx.x;
  if (i >= B * 64) return;
  int b = i >> 6, j = i & 63;
  s[i] = (j < 32) ? (_Float16)z[b * 32 + j] : (_Float16)0.0f;
  if (j < 32) c[b * 32 + j] = 0.0f;
}

// torch gate order i,f,g,o ; writes h into both input and hidden slots of s
__global__ void k_lstm_cell(const float* __restrict__ gates, float* __restrict__ c,
                            _Float16* __restrict__ s, _Float16* __restrict__ hb, int B) {
  int idx = blockIdx.x * blockDim.x + threadIdx.x;
  if (idx >= B * 32) return;
  int b = idx >> 5, j = idx & 31;
  const float* g = gates + (size_t)b * 128;
  float ig = sigmf(g[j]);
  float fg = sigmf(g[32 + j]);
  float gg = tanhf(g[64 + j]);
  float og = sigmf(g[96 + j]);
  float cn = fg * c[idx] + ig * gg;
  float hn = og * tanhf(cn);
  c[idx] = cn;
  _Float16 hh = (_Float16)hn;
  s[(size_t)b * 64 + j] = hh;
  s[(size_t)b * 64 + 32 + j] = hh;
  hb[idx] = hh;
}

__global__ void k_head_out(const _Float16* __restrict__ h2, const float* __restrict__ w3,
                           const float* __restrict__ b3, float* __restrict__ out,
                           int B, int t) {
  int b = blockIdx.x * blockDim.x + threadIdx.x;
  if (b >= B) return;
  float acc = b3[0];
#pragma unroll
  for (int j = 0; j < 32; ++j) acc += (float)h2[b * 32 + j] * w3[j];
  out[b * 25 + t] = sigmf(acc);
}

// ---------------------------------------------------------------------------

extern "C" void kernel_launch(void* const* d_in, const int* in_sizes, int n_in,
                              void* d_out, int out_size, void* d_ws, size_t ws_size,
                              hipStream_t stream) {
  (void)in_sizes; (void)n_in; (void)out_size; (void)ws_size;
  auto P = [&](int i) { return (const float*)d_in[i]; };
  const float* x = P(0);
  // Input order (setup_inputs dict insertion order):
  // 0:x ; convs i=0..5 -> w=1+4i,b=2+4i,g=3+4i,be=4+4i ; conv_f 25..28 ;
  // fc1 29..32 ; fc2 33..34 ; enc 35..44 ; head 45..54 ; lstm 55..58

  // ---- workspace carve (all written before read) ----
  size_t pos = 0;
  auto carve = [&](size_t bytes) {
    void* p = (char*)d_ws + pos;
    pos += (bytes + 255) & ~(size_t)255;
    return p;
  };
  float*    PRE   = (float*)carve((size_t)512 * 64 * 64 * 4 * sizeof(float));     // max preact
  _Float16* ACT   = (_Float16*)carve((size_t)512 * 32 * 32 * 4 * sizeof(_Float16)); // max act
  _Float16* COLS  = (_Float16*)carve((size_t)131072 * 96 * sizeof(_Float16));     // im2col
  float*    MEAN  = (float*)carve(128 * sizeof(float));
  float*    ISTD  = (float*)carve(128 * sizeof(float));
  _Float16* Wc3   = (_Float16*)carve(16 * 96 * 2);
  _Float16* Wc4   = (_Float16*)carve(32 * 160 * 2);
  _Float16* Wc5   = (_Float16*)carve(64 * 288 * 2);
  _Float16* Wc6   = (_Float16*)carve(128 * 576 * 2);
  _Float16* Wcf   = (_Float16*)carve(128 * 128 * 2);
  _Float16* Wfc1  = (_Float16*)carve(128 * 128 * 2);
  _Float16* Wfc2  = (_Float16*)carve(64 * 128 * 2);
  _Float16* We1   = (_Float16*)carve(32 * 192 * 2);
  _Float16* We2   = (_Float16*)carve(32 * 32 * 2);
  _Float16* We3   = (_Float16*)carve(32 * 32 * 2);
  _Float16* Wh1   = (_Float16*)carve(32 * 32 * 2);
  _Float16* Wh2   = (_Float16*)carve(32 * 32 * 2);
  _Float16* WLC   = (_Float16*)carve(128 * 64 * 2);
  float*    BSUM  = (float*)carve(128 * sizeof(float));
  _Float16* MERGED= (_Float16*)carve((size_t)512 * 192 * 2);
  _Float16* S     = (_Float16*)carve((size_t)512 * 64 * 2);
  float*    CST   = (float*)carve((size_t)512 * 32 * sizeof(float));
  _Float16* HB    = (_Float16*)carve((size_t)512 * 32 * 2);
  float*    GATES = (float*)carve((size_t)512 * 128 * sizeof(float));

  auto nb = [](long long n, int t) { return (unsigned)((n + t - 1) / t); };
  auto gemm = [&](const _Float16* A, const _Float16* Wt, const float* bias, float* Y,
                  int M, int N, int K) {
    int tiles = (M >> 4) * (N >> 4);
    k_gemm_wmma<<<nb(tiles, 4), 128, 0, stream>>>(A, Wt, bias, Y, M, N, K, tiles);
  };

  // ---- weight prep (f32 -> f16, repack/pad K to mult of 32) ----
  k_cast_f16<<<nb(128 * 128, 256), 256, 0, stream>>>(P(29), Wfc1, 128 * 128);
  k_cast_f16<<<nb(64 * 128, 256), 256, 0, stream>>>(P(33), Wfc2, 64 * 128);
  k_cast_f16<<<nb(32 * 192, 256), 256, 0, stream>>>(P(35), We1, 32 * 192);
  k_cast_f16<<<nb(1024, 256), 256, 0, stream>>>(P(39), We2, 1024);
  k_cast_f16<<<nb(1024, 256), 256, 0, stream>>>(P(43), We3, 1024);
  k_cast_f16<<<nb(1024, 256), 256, 0, stream>>>(P(45), Wh1, 1024);
  k_cast_f16<<<nb(1024, 256), 256, 0, stream>>>(P(49), Wh2, 1024);
  k_repack_conv_w<<<nb(16 * 96, 256), 256, 0, stream>>>(P(1 + 4 * 2), Wc3, 8, 16, 96);
  k_repack_conv_w<<<nb(32 * 160, 256), 256, 0, stream>>>(P(1 + 4 * 3), Wc4, 16, 32, 160);
  k_repack_conv_w<<<nb(64 * 288, 256), 256, 0, stream>>>(P(1 + 4 * 4), Wc5, 32, 64, 288);
  k_repack_conv_w<<<nb(128 * 576, 256), 256, 0, stream>>>(P(1 + 4 * 5), Wc6, 64, 128, 576);
  k_repack_convf<<<nb(128 * 128, 256), 256, 0, stream>>>(P(25), Wcf);
  k_pack_lstm<<<nb(128 * 64, 256), 256, 0, stream>>>(P(55), P(56), P(57), P(58), WLC, BSUM);

  // ---- per-frame encoder (BN stats are per frame -> 3 independent passes) ----
  for (int t = 0; t < 3; ++t) {
    // L1 (3->4) direct
    k_conv1<<<nb(512LL * 64 * 64, 256), 256, 0, stream>>>(x, P(1), P(2), PRE, t);
    k_bn_stats<<<4, 256, 0, stream>>>(PRE, MEAN, ISTD, 4, 512 * 64 * 64);
    k_bn_relu_pool<<<nb(512LL * 32 * 32 * 4, 256), 256, 0, stream>>>(
        PRE, MEAN, ISTD, P(3), P(4), ACT, 512, 64, 64, 4);
    // L2 (4->8) direct
    k_conv_direct<<<nb(512LL * 32 * 32, 256), 256, 0, stream>>>(
        ACT, P(5), P(6), PRE, 512, 32, 32, 4, 8);
    k_bn_stats<<<8, 256, 0, stream>>>(PRE, MEAN, ISTD, 8, 512 * 32 * 32);
    k_bn_relu_pool<<<nb(512LL * 16 * 16 * 8, 256), 256, 0, stream>>>(
        PRE, MEAN, ISTD, P(7), P(8), ACT, 512, 32, 32, 8);
    // L3..L6 via im2col + WMMA implicit GEMM
    struct LayerDesc { int H, Ci, Co, Kp; _Float16* Wp; };
    LayerDesc Ld[4] = {{16, 8, 16, 96, Wc3}, {8, 16, 32, 160, Wc4},
                       {4, 32, 64, 288, Wc5}, {2, 64, 128, 576, Wc6}};
    for (int li = 0; li < 4; ++li) {
      LayerDesc L = Ld[li];
      long long Mr = 512LL * L.H * L.H;
      int pi = 1 + 4 * (2 + li);
      k_im2col3x3<<<nb(Mr * L.Kp, 256), 256, 0, stream>>>(ACT, COLS, 512, L.H, L.H, L.Ci, L.Kp);
      gemm(COLS, L.Wp, P(pi + 1), PRE, (int)Mr, L.Co, L.Kp);
      k_bn_stats<<<L.Co, 256, 0, stream>>>(PRE, MEAN, ISTD, L.Co, (int)Mr);
      k_bn_relu_pool<<<nb((Mr / 4) * L.Co, 256), 256, 0, stream>>>(
          PRE, MEAN, ISTD, P(pi + 2), P(pi + 3), ACT, 512, L.H, L.H, L.Co);
    }
    // conv_f at 1x1 spatial == 128x128 matmul with center tap
    gemm(ACT, Wcf, P(26), PRE, 512, 128, 128);
    k_bn_stats<<<128, 256, 0, stream>>>(PRE, MEAN, ISTD, 128, 512);
    k_bn_relu<<<nb(512 * 128, 256), 256, 0, stream>>>(PRE, MEAN, ISTD, P(27), P(28), ACT,
                                                      128, 512LL * 128);
    // fc1 + BN + relu
    gemm(ACT, Wfc1, P(30), PRE, 512, 128, 128);
    k_bn_stats<<<128, 256, 0, stream>>>(PRE, MEAN, ISTD, 128, 512);
    k_bn_relu<<<nb(512 * 128, 256), 256, 0, stream>>>(PRE, MEAN, ISTD, P(31), P(32), ACT,
                                                      128, 512LL * 128);
    // fc2 -> emb slice of merged[512,192]
    gemm(ACT, Wfc2, P(34), PRE, 512, 64, 128);
    k_cast_scatter<<<nb(512 * 64, 256), 256, 0, stream>>>(PRE, MERGED, 512, 64, 192, t * 64);
  }

  // ---- enc MLP3: 192 -> 32 -> 32 ----
  gemm(MERGED, We1, P(36), PRE, 512, 32, 192);
  k_bn_stats<<<32, 256, 0, stream>>>(PRE, MEAN, ISTD, 32, 512);
  k_bn_relu<<<nb(512 * 32, 256), 256, 0, stream>>>(PRE, MEAN, ISTD, P(37), P(38), ACT,
                                                   32, 512LL * 32);
  gemm(ACT, We2, P(40), PRE, 512, 32, 32);
  k_bn_stats<<<32, 256, 0, stream>>>(PRE, MEAN, ISTD, 32, 512);
  k_bn_relu<<<nb(512 * 32, 256), 256, 0, stream>>>(PRE, MEAN, ISTD, P(41), P(42), ACT,
                                                   32, 512LL * 32);
  gemm(ACT, We3, P(44), PRE, 512, 32, 32);  // z (f32) in PRE
  k_lstm_init<<<nb(512 * 64, 256), 256, 0, stream>>>(PRE, S, CST, 512);

  // ---- 25 LSTM steps: gates = [inp|h] @ [Wih|Whh]^T + (bih+bhh) ----
  for (int t = 0; t < 25; ++t) {
    gemm(S, WLC, BSUM, GATES, 512, 128, 64);
    k_lstm_cell<<<nb(512 * 32, 256), 256, 0, stream>>>(GATES, CST, S, HB, 512);
    // head MLP3 with per-step batch BN
    gemm(HB, Wh1, P(46), PRE, 512, 32, 32);
    k_bn_stats<<<32, 256, 0, stream>>>(PRE, MEAN, ISTD, 32, 512);
    k_bn_relu<<<nb(512 * 32, 256), 256, 0, stream>>>(PRE, MEAN, ISTD, P(47), P(48), ACT,
                                                     32, 512LL * 32);
    gemm(ACT, Wh2, P(50), PRE, 512, 32, 32);
    k_bn_stats<<<32, 256, 0, stream>>>(PRE, MEAN, ISTD, 32, 512);
    k_bn_relu<<<nb(512 * 32, 256), 256, 0, stream>>>(PRE, MEAN, ISTD, P(51), P(52), ACT,
                                                     32, 512LL * 32);
    k_head_out<<<nb(512, 256), 256, 0, stream>>>(ACT, P(53), P(54), (float*)d_out, 512, t);
  }
}
